// MambaTransfomer_77378130805215
// MI455X (gfx1250) — matrix operand order repaired and checked by reading, not compiled, not run on hardware
//
#include <hip/hip_runtime.h>
#include <hip/hip_bf16.h>

typedef __attribute__((ext_vector_type(16))) _Float16 v16h;
typedef __attribute__((ext_vector_type(8)))  _Float16 v8h;
typedef __attribute__((ext_vector_type(8)))  float    v8f;
typedef int v4i_t __attribute__((vector_size(4 * sizeof(int))));

#define DEV __device__ __forceinline__
#define GAS __attribute__((address_space(1)))
#define LAS __attribute__((address_space(3)))

constexpr int S_   = 8192;  // L*H*W
constexpr int DM   = 256;   // d_model
constexpr int DI   = 256;   // d_inner
constexpr int DS   = 8;     // d_state
constexpr int DR   = 16;    // dt_rank
constexpr int DBLC = 32;    // dt_rank + 2*d_state
constexpr int DC   = 128;   // d_conv1d
constexpr int NCH  = 64;    // scan chunks
constexpr int CLEN = 128;   // chunk length (NCH*CLEN == S_)

// ---------------- async global->LDS copy (16 bytes per call) -----------------

#if defined(__has_builtin)
#  if __has_builtin(__builtin_amdgcn_global_load_async_to_lds_b128)
#    define ASYNC_LDS_BUILTIN 1
#  endif
#  if __has_builtin(__builtin_amdgcn_s_wait_asynccnt)
#    define WAIT_ASYNC_BUILTIN 1
#  endif
#endif

DEV void async_copy16(const void* g, void* l) {
#ifdef ASYNC_LDS_BUILTIN
  __builtin_amdgcn_global_load_async_to_lds_b128(
      (GAS v4i_t*)(unsigned long long)g,
      (LAS v4i_t*)(unsigned)(unsigned long long)l, 0, 0);
#else
  asm volatile("global_load_async_to_lds_b128 %0, %1, off"
               :: "v"((unsigned)(unsigned long long)l), "v"(g)
               : "memory");
#endif
}

DEV void async_wait0() {
#ifdef WAIT_ASYNC_BUILTIN
  __builtin_amdgcn_s_wait_asynccnt(0);
#else
  asm volatile("s_wait_asynccnt 0x0" ::: "memory");
#endif
}

// ---------------- WMMA fragment helpers (CDNA5 16x16x32 f16 layouts) ---------

DEV v16h load_a16(const _Float16* __restrict__ A, int lda, int k0) {
  const int lane = threadIdx.x & 31;
  const int hh = lane >> 4;
  const int m  = lane & 15;
  const _Float16* p = A + (size_t)m * lda + k0 + hh * 8;
  const v8h lo = *(const v8h*)(p);
  const v8h hi = *(const v8h*)(p + 16);
  v16h r;
#pragma unroll
  for (int e = 0; e < 8; ++e) { r[e] = lo[e]; r[e + 8] = hi[e]; }
  return r;
}

// B fragment from a [N][K] row-major tile (generic pointer; LDS tiles infer ds ops)
DEV v16h load_b16(const _Float16* __restrict__ Wt, int ldw, int k0) {
  const int lane = threadIdx.x & 31;
  const int hh = lane >> 4;
  const int n  = lane & 15;
  const _Float16* p = Wt + (size_t)n * ldw + k0 + hh * 8;
  const v8h lo = *(const v8h*)(p);
  const v8h hi = *(const v8h*)(p + 16);
  v16h r;
#pragma unroll
  for (int e = 0; e < 8; ++e) { r[e] = lo[e]; r[e + 8] = hi[e]; }
  return r;
}

DEV void store_c16(float* __restrict__ C, int ldc, const v8f& c) {
  const int lane = threadIdx.x & 31;
  const int hh = lane >> 4;
  const int n  = lane & 15;
#pragma unroll
  for (int r = 0; r < 8; ++r) C[(size_t)(r + hh * 8) * ldc + n] = c[r];
}

DEV float silu_f(float x) { return x / (1.0f + __expf(-x)); }
DEV float softplus_f(float x) { return (x > 20.0f) ? x : log1pf(__expf(x)); }

// ---------------- small prep kernels -----------------------------------------

__global__ void k_f32_to_f16(const float* __restrict__ src, _Float16* __restrict__ dst, int n) {
  int i = blockIdx.x * blockDim.x + threadIdx.x;
  if (i < n) dst[i] = (_Float16)src[i];
}

__global__ void k_fuse_w(const float* __restrict__ op, const float* __restrict__ sow,
                         const float* __restrict__ tow,
                         _Float16* __restrict__ W1, _Float16* __restrict__ W2,
                         _Float16* __restrict__ W3) {
  const int e = blockIdx.x;
  const int d = threadIdx.x;
  float a1 = 0.f, a2 = 0.f;
#pragma unroll 4
  for (int k = 0; k < 64; ++k) {
    a1 += op[e * DM + k]      * sow[k * DI + d];
    a2 += op[e * DM + 64 + k] * tow[k * DI + d];
  }
  W1[e * DI + d] = (_Float16)a1;
  W2[e * DI + d] = (_Float16)a2;
  if (d < DC) W3[e * DC + d] = (_Float16)op[e * DM + 128 + d];
}

// ---------------- WMMA GEMM with async-LDS weight staging ---------------------
// C[M][N] = A[M][K] * Wt[N][K]^T. One block = 8 waves, 8 consecutive M-tiles,
// sharing one N-strip of 16*NT columns whose weights live in LDS.

template <int NT>
__global__ void k_gemm(const _Float16* __restrict__ A, const _Float16* __restrict__ Wt,
                       float* __restrict__ C, int M, int N, int K) {
  extern __shared__ _Float16 sB[];                 // [16*NT][K]
  const int nstrips = N / (16 * NT);
  const int bm = blockIdx.x / nstrips;
  const int nb = blockIdx.x % nstrips;
  const int col0 = nb * 16 * NT;

  {  // cooperative async stage of the weight strip
    const _Float16* src = Wt + (size_t)col0 * K;
    const int chunks = (16 * NT * K) / 8;          // 16-byte chunks
    for (int c = threadIdx.x; c < chunks; c += 256)
      async_copy16(src + (size_t)c * 8, sB + (size_t)c * 8);
    async_wait0();
    __syncthreads();
  }

  const int wave = threadIdx.x >> 5;
  const int lane = threadIdx.x & 31;
  const int row0 = (bm * 8 + wave) * 16;
  const _Float16* Arow = A + (size_t)row0 * K;
  v8f acc[NT] = {};
  for (int k0 = 0; k0 < K; k0 += 32) {
    __builtin_prefetch(Arow + (size_t)(lane & 15) * K + k0 + 64, 0, 1);
    v16h a = load_a16(Arow, K, k0);
#pragma unroll
    for (int t = 0; t < NT; ++t) {
      v16h b = load_b16(sB + (size_t)(t * 16) * K, K, k0);
      acc[t] = __builtin_amdgcn_wmma_f32_16x16x32_f16(
          false, a, false, b, (short)0, acc[t], false, false);
    }
  }
#pragma unroll
  for (int t = 0; t < NT; ++t)
    store_c16(C + (size_t)row0 * N + col0 + t * 16, N, acc[t]);
}

// ---------------- depthwise conv + SiLU (SSM branches) ------------------------

__global__ void k_conv_ssm(const float* __restrict__ xi, const float* __restrict__ cw,
                           const float* __restrict__ cb, _Float16* __restrict__ xc,
                           int timeMode) {
  const int d = threadIdx.x;      // 256
  const int l = blockIdx.x;       // 8192
  float acc = cb[d];
#pragma unroll
  for (int k = 0; k < 3; ++k) {
    const int ll = l + k - 1;
    if (ll >= 0 && ll < S_) {
      const int sr = timeMode ? ((ll & 7) * 1024 + (ll >> 3)) : ll;
      acc += cw[d * 3 + k] * xi[(size_t)sr * DI + d];
    }
  }
  xc[(size_t)l * DI + d] = (_Float16)silu_f(acc);
}

__global__ void k_conv_branch(const float* __restrict__ xi, const float* __restrict__ bw,
                              const float* __restrict__ bb, _Float16* __restrict__ cy) {
  const int c = threadIdx.x;      // 128
  const int s = blockIdx.x;       // 8192
  float acc = bb[c];
#pragma unroll
  for (int k = 0; k < 3; ++k) {
    const int ss = s + k - 1;
    if (ss >= 0 && ss < S_) {
      acc += bw[c * 6 + k]     * xi[(size_t)ss * DI + 2 * c];
      acc += bw[c * 6 + 3 + k] * xi[(size_t)ss * DI + 2 * c + 1];
    }
  }
  cy[(size_t)s * DC + c] = (_Float16)silu_f(acc);
}

// ---------------- chunked selective scan --------------------------------------
// Block = (chunk, 32-channel slab). Whole chunk's dt/B/C rows (16 KB) and the
// xc slab (8 KB) are async-staged to LDS once; 128 recurrence steps then run
// with zero barriers. lane=(d,n): 8 states per channel.

DEV void stage_scan_tiles(const float* __restrict__ dblT, const _Float16* __restrict__ xc,
                          int l0, int d0, float* sdbl, _Float16* sxc) {
  const float* srcd = dblT + (size_t)l0 * DBLC;
  for (int c = threadIdx.x; c < (CLEN * DBLC) / 4; c += 256)        // f32 x4 = 16B
    async_copy16(srcd + (size_t)c * 4, sdbl + (size_t)c * 4);
  for (int r = threadIdx.x; r < CLEN * 4; r += 256) {               // 4x16B per row
    const int row = r >> 2, cc = r & 3;
    async_copy16(xc + (size_t)(l0 + row) * DI + d0 + cc * 8,
                 sxc + row * 32 + cc * 8);
  }
  async_wait0();
  __syncthreads();
}

__global__ void k_scan_p1(const float* __restrict__ dblT, const _Float16* __restrict__ xc,
                          const float* __restrict__ dtw, const float* __restrict__ dtb,
                          const float* __restrict__ Alog,
                          float* __restrict__ prodA, float* __restrict__ hend) {
  __shared__ float    sdbl[CLEN * DBLC];
  __shared__ _Float16 sxc[CLEN * 32];
  const int tid = threadIdx.x;
  const int chunk = blockIdx.x >> 3;
  const int d0 = (blockIdx.x & 7) << 5;
  const int l0 = chunk * CLEN;
  stage_scan_tiles(dblT, xc, l0, d0, sdbl, sxc);

  const int d = d0 + (tid >> 3);
  const int n = tid & 7;
  const float Ac  = -__expf(Alog[d * DS + n]);
  const float tb2 = 2.0f * dtb[d];
  float wdt[DR];
#pragma unroll
  for (int r = 0; r < DR; ++r) wdt[r] = dtw[d * DR + r];
  float h = 0.0f, p = 1.0f;
  for (int t = 0; t < CLEN; ++t) {
    const float* row = sdbl + t * DBLC;
    float dot = tb2;
#pragma unroll
    for (int r = 0; r < DR; ++r) dot += wdt[r] * row[r];
    const float delta = softplus_f(dot);
    const float dA = __expf(delta * Ac);
    const float u  = (float)sxc[t * 32 + (d & 31)];
    h = dA * h + delta * row[DR + n] * u;
    p *= dA;
  }
  const size_t idx = ((size_t)chunk * DI + d) * DS + n;
  prodA[idx] = p;
  hend[idx]  = h;
}

__global__ void k_scan_p2(const float* __restrict__ prodA, const float* __restrict__ hend,
                          float* __restrict__ h0) {
  const int g = blockIdx.x * 1024 + threadIdx.x;   // (d*8+n) in [0,2048)
  float h = 0.0f;
  for (int c = 0; c < NCH; ++c) {
    const size_t idx = (size_t)c * (DI * DS) + g;
    h0[idx] = h;
    h = prodA[idx] * h + hend[idx];
  }
}

__global__ void k_scan_p3(const float* __restrict__ dblT, const _Float16* __restrict__ xc,
                          const float* __restrict__ dtw, const float* __restrict__ dtb,
                          const float* __restrict__ Alog, const float* __restrict__ h0,
                          const float* __restrict__ Dp, _Float16* __restrict__ yh,
                          int timeMode) {
  __shared__ float    sdbl[CLEN * DBLC];
  __shared__ _Float16 sxc[CLEN * 32];
  const int tid = threadIdx.x;
  const int chunk = blockIdx.x >> 3;
  const int d0 = (blockIdx.x & 7) << 5;
  const int l0 = chunk * CLEN;
  stage_scan_tiles(dblT, xc, l0, d0, sdbl, sxc);

  const int d = d0 + (tid >> 3);
  const int n = tid & 7;
  const float Ac  = -__expf(Alog[d * DS + n]);
  const float tb2 = 2.0f * dtb[d];
  const float Dd  = Dp[d];
  float wdt[DR];
#pragma unroll
  for (int r = 0; r < DR; ++r) wdt[r] = dtw[d * DR + r];
  float h = h0[((size_t)chunk * DI + d) * DS + n];
  for (int t = 0; t < CLEN; ++t) {
    const int l = l0 + t;
    const float* row = sdbl + t * DBLC;
    float dot = tb2;
#pragma unroll
    for (int r = 0; r < DR; ++r) dot += wdt[r] * row[r];
    const float delta = softplus_f(dot);
    const float dA = __expf(delta * Ac);
    const float u  = (float)sxc[t * 32 + (d & 31)];
    h = dA * h + delta * row[DR + n] * u;
    float v = h * row[DR + DS + n];             // h * C[n,l]
    v += __shfl_xor(v, 1, 8);                   // 8-state reduce (wave32 permutes)
    v += __shfl_xor(v, 2, 8);
    v += __shfl_xor(v, 4, 8);
    if (n == 0) {
      const float y = v + u * Dd;
      const int dst = timeMode ? ((l & 7) * 1024 + (l >> 3)) : l;
      yh[(size_t)dst * DI + d] = (_Float16)y;
    }
  }
}

// ---------------- fused output GEMM: K = 256 (ys) + 256 (yt) + 128 (cy) -------
// One 32 KB LDS buffer reused per weight segment.

__global__ void k_final(const _Float16* __restrict__ Ys, const _Float16* __restrict__ Yt,
                        const _Float16* __restrict__ Cy, const _Float16* __restrict__ W1,
                        const _Float16* __restrict__ W2, const _Float16* __restrict__ W3,
                        float* __restrict__ out) {
  extern __shared__ _Float16 sW[];                // 64 x 256 max
  const int bm = blockIdx.x >> 2;
  const int nb = blockIdx.x & 3;
  const int col0 = nb * 64;
  const int wave = threadIdx.x >> 5;
  const int lane = threadIdx.x & 31;
  const int row0 = (bm * 8 + wave) * 16;
  v8f acc[4] = {};

  // --- segment 1: Ys * W1^T (K=256) ---
  {
    const _Float16* src = W1 + (size_t)col0 * DI;
    for (int c = threadIdx.x; c < (64 * DI) / 8; c += 256)
      async_copy16(src + (size_t)c * 8, sW + (size_t)c * 8);
    async_wait0();
    __syncthreads();
  }
  const _Float16* As = Ys + (size_t)row0 * DI;
  for (int k0 = 0; k0 < DI; k0 += 32) {
    __builtin_prefetch(As + (size_t)(lane & 15) * DI + k0 + 64, 0, 1);
    v16h a = load_a16(As, DI, k0);
#pragma unroll
    for (int t = 0; t < 4; ++t) {
      v16h b = load_b16(sW + (size_t)(t * 16) * DI, DI, k0);
      acc[t] = __builtin_amdgcn_wmma_f32_16x16x32_f16(false, a, false, b, (short)0, acc[t], false, false);
    }
  }
  // --- segment 2: Yt * W2^T (K=256) ---
  __syncthreads();
  {
    const _Float16* src = W2 + (size_t)col0 * DI;
    for (int c = threadIdx.x; c < (64 * DI) / 8; c += 256)
      async_copy16(src + (size_t)c * 8, sW + (size_t)c * 8);
    async_wait0();
    __syncthreads();
  }
  const _Float16* At = Yt + (size_t)row0 * DI;
  for (int k0 = 0; k0 < DI; k0 += 32) {
    __builtin_prefetch(At + (size_t)(lane & 15) * DI + k0 + 64, 0, 1);
    v16h a = load_a16(At, DI, k0);
#pragma unroll
    for (int t = 0; t < 4; ++t) {
      v16h b = load_b16(sW + (size_t)(t * 16) * DI, DI, k0);
      acc[t] = __builtin_amdgcn_wmma_f32_16x16x32_f16(false, a, false, b, (short)0, acc[t], false, false);
    }
  }
  // --- segment 3: Cy * W3^T (K=128) ---
  __syncthreads();
  {
    const _Float16* src = W3 + (size_t)col0 * DC;
    for (int c = threadIdx.x; c < (64 * DC) / 8; c += 256)
      async_copy16(src + (size_t)c * 8, sW + (size_t)c * 8);
    async_wait0();
    __syncthreads();
  }
  const _Float16* Ac = Cy + (size_t)row0 * DC;
  for (int k0 = 0; k0 < DC; k0 += 32) {
    v16h a = load_a16(Ac, DC, k0);
#pragma unroll
    for (int t = 0; t < 4; ++t) {
      v16h b = load_b16(sW + (size_t)(t * 16) * DC, DC, k0);
      acc[t] = __builtin_amdgcn_wmma_f32_16x16x32_f16(false, a, false, b, (short)0, acc[t], false, false);
    }
  }
#pragma unroll
  for (int t = 0; t < 4; ++t)
    store_c16(out + (size_t)row0 * DM + col0 + t * 16, DM, acc[t]);
}

// ---------------- host orchestration ------------------------------------------

extern "C" void kernel_launch(void* const* d_in, const int* in_sizes, int n_in,
                              void* d_out, int out_size, void* d_ws, size_t ws_size,
                              hipStream_t stream) {
  (void)in_sizes; (void)n_in; (void)out_size; (void)ws_size;
  const float* x    = (const float*)d_in[0];
  const float* inw  = (const float*)d_in[1];
  const float* scw  = (const float*)d_in[2];
  const float* scb  = (const float*)d_in[3];
  const float* sxw  = (const float*)d_in[4];
  const float* sdtw = (const float*)d_in[5];
  const float* sdtb = (const float*)d_in[6];
  const float* sAl  = (const float*)d_in[7];
  const float* sD   = (const float*)d_in[8];
  const float* sow  = (const float*)d_in[9];
  const float* tcw  = (const float*)d_in[10];
  const float* tcb  = (const float*)d_in[11];
  const float* txw  = (const float*)d_in[12];
  const float* tdtw = (const float*)d_in[13];
  const float* tdtb = (const float*)d_in[14];
  const float* tAl  = (const float*)d_in[15];
  const float* tD   = (const float*)d_in[16];
  const float* tow  = (const float*)d_in[17];
  const float* bcw  = (const float*)d_in[18];
  const float* bcb  = (const float*)d_in[19];
  const float* opw  = (const float*)d_in[20];
  float* out = (float*)d_out;

  char* base = (char*)d_ws;
  size_t off = 0;
  auto alloc = [&](size_t bytes) -> void* {
    off = (off + 255) & ~(size_t)255;
    void* p = base + off;
    off += bytes;
    return p;
  };

  _Float16* xh   = (_Float16*)alloc((size_t)S_ * DM * 2);
  float*    xi   = (float*)   alloc((size_t)S_ * DI * 4);
  _Float16* inwh = (_Float16*)alloc((size_t)DI * DM * 2);
  _Float16* sxwh = (_Float16*)alloc((size_t)DBLC * DI * 2);
  _Float16* txwh = (_Float16*)alloc((size_t)DBLC * DI * 2);
  _Float16* xcs  = (_Float16*)alloc((size_t)S_ * DI * 2);
  _Float16* xct  = (_Float16*)alloc((size_t)S_ * DI * 2);
  float*    dbls = (float*)   alloc((size_t)S_ * DBLC * 4);
  float*    dblt = (float*)   alloc((size_t)S_ * DBLC * 4);
  float*    pAs  = (float*)   alloc((size_t)NCH * DI * DS * 4);
  float*    hes  = (float*)   alloc((size_t)NCH * DI * DS * 4);
  float*    h0s  = (float*)   alloc((size_t)NCH * DI * DS * 4);
  float*    pAt  = (float*)   alloc((size_t)NCH * DI * DS * 4);
  float*    het  = (float*)   alloc((size_t)NCH * DI * DS * 4);
  float*    h0t  = (float*)   alloc((size_t)NCH * DI * DS * 4);
  _Float16* ysh  = (_Float16*)alloc((size_t)S_ * DI * 2);
  _Float16* yth  = (_Float16*)alloc((size_t)S_ * DI * 2);
  _Float16* cyh  = (_Float16*)alloc((size_t)S_ * DC * 2);
  _Float16* W1h  = (_Float16*)alloc((size_t)DM * DI * 2);
  _Float16* W2h  = (_Float16*)alloc((size_t)DM * DI * 2);
  _Float16* W3h  = (_Float16*)alloc((size_t)DM * DC * 2);

  // 1) precision prep + fused weights
  k_f32_to_f16<<<(S_ * DM + 255) / 256, 256, 0, stream>>>(x, xh, S_ * DM);
  k_f32_to_f16<<<(DI * DM + 255) / 256, 256, 0, stream>>>(inw, inwh, DI * DM);
  k_f32_to_f16<<<(DBLC * DI + 255) / 256, 256, 0, stream>>>(sxw, sxwh, DBLC * DI);
  k_f32_to_f16<<<(DBLC * DI + 255) / 256, 256, 0, stream>>>(txw, txwh, DBLC * DI);
  k_fuse_w<<<DM, 256, 0, stream>>>(opw, sow, tow, W1h, W2h, W3h);

  // 2) in_proj GEMM (WMMA, async-LDS weights): 64 M-blocks x 4 N-strips
  k_gemm<4><<<(S_ / 128) * (DI / 64), 256, 16 * 4 * DM * 2, stream>>>(xh, inwh, xi, S_, DI, DM);

  // 3) depthwise convs + SiLU and grouped conv branch
  k_conv_ssm<<<S_, DI, 0, stream>>>(xi, scw, scb, xcs, 0);
  k_conv_ssm<<<S_, DI, 0, stream>>>(xi, tcw, tcb, xct, 1);
  k_conv_branch<<<S_, DC, 0, stream>>>(xi, bcw, bcb, cyh);

  // 4) x_proj GEMMs (WMMA): N=32 -> single strip per block
  k_gemm<2><<<S_ / 128, 256, 16 * 2 * DI * 2, stream>>>(xcs, sxwh, dbls, S_, DBLC, DI);
  k_gemm<2><<<S_ / 128, 256, 16 * 2 * DI * 2, stream>>>(xct, txwh, dblt, S_, DBLC, DI);

  // 5) chunk-parallel selective scans (async-LDS staged, barrier-free inner loop)
  k_scan_p1<<<NCH * 8, 256, 0, stream>>>(dbls, xcs, sdtw, sdtb, sAl, pAs, hes);
  k_scan_p1<<<NCH * 8, 256, 0, stream>>>(dblt, xct, tdtw, tdtb, tAl, pAt, het);
  k_scan_p2<<<2, 1024, 0, stream>>>(pAs, hes, h0s);
  k_scan_p2<<<2, 1024, 0, stream>>>(pAt, het, h0t);
  k_scan_p3<<<NCH * 8, 256, 0, stream>>>(dbls, xcs, sdtw, sdtb, sAl, h0s, sD, ysh, 0);
  k_scan_p3<<<NCH * 8, 256, 0, stream>>>(dblt, xct, tdtw, tdtb, tAl, h0t, tD, yth, 1);

  // 6) fused output GEMM (sy/ty/cy projections + out_proj collapsed; WMMA)
  k_final<<<(S_ / 128) * (DM / 64), 256, 64 * DI * 2, stream>>>(ysh, yth, cyh, W1h, W2h, W3h, out);
}
